// GroupQueryAttention_36859409334432
// MI455X (gfx1250) — matrix operand order repaired and checked by reading, not compiled, not run on hardware
//
#include <hip/hip_runtime.h>

// ---------------------------------------------------------------------------
// GQA forward for MI455X (gfx1250): QKV projection + causal flash attention.
// All matrix math in bf16 WMMA (v_wmma_f32_16x16x32_bf16) with fp32 accum.
// Attention K-tiles staged with CDNA5 async copies (global_load_async_to_lds),
// double-buffered, one barrier per key tile.
// B=1, S=4096, D=1024, H=8, hd=128.
// ---------------------------------------------------------------------------

#define S_LEN 4096
#define D_DIM 1024
#define NH    8
#define HD    128

typedef __attribute__((ext_vector_type(8)))  __bf16 v8bf;
typedef __attribute__((ext_vector_type(16))) __bf16 v16bf;
typedef __attribute__((ext_vector_type(8)))  float  v8f;

// D = A(16x32) * B(32x16) + C, bf16 in / f32 out, one WMMA per wave.
__device__ __forceinline__ v8f wmma_bf16(v16bf a, v16bf b, v8f c) {
  return __builtin_amdgcn_wmma_f32_16x16x32_bf16(
      /*neg_a=*/false, a, /*neg_b=*/false, b,
      /*c_mod=*/(short)0, c, /*reuse_a=*/false, /*reuse_b=*/false);
}

// A fragment (16x32, bf16), element (m,k) at src[m*stride + k].
// Lane L -> m = L&15; lanes 0-15 hold k = 0..7 & 16..23,
// lanes 16-31 hold k = 8..15 & 24..31 (2 bf16 per VGPR).
__device__ __forceinline__ v16bf frag_a(const __bf16* base, int stride) {
  const int lane = threadIdx.x & 31;
  const int m    = lane & 15;
  const int koff = (lane >> 4) << 3;          // 0 or 8
  const __bf16* p = base + m * stride + koff;
  v8bf lo = *(const v8bf*)(p);
  v8bf hi = *(const v8bf*)(p + 16);
  v16bf f;
#pragma unroll
  for (int j = 0; j < 8; ++j) { f[j] = lo[j]; f[8 + j] = hi[j]; }
  return f;
}

// B fragment (32x16, bf16) from a transposed source: element (k,n) at
// src[n*stride + k].  Lane L -> n = L&15; lanes 0-15 hold k = 0..7 & 16..23,
// lanes 16-31 hold k = 8..15 & 24..31.
__device__ __forceinline__ v16bf frag_bt(const __bf16* base, int stride) {
  const int lane = threadIdx.x & 31;
  const int n  = lane & 15;
  const int k0 = (lane >> 4) << 3;            // 0 or 8
  const __bf16* p = base + n * stride + k0;
  v8bf lo = *(const v8bf*)(p);
  v8bf hi = *(const v8bf*)(p + 16);
  v16bf f;
#pragma unroll
  for (int j = 0; j < 8; ++j) { f[j] = lo[j]; f[8 + j] = hi[j]; }
  return f;
}

// CDNA5 async copy: 16 bytes per lane, global -> LDS, tracked by ASYNCcnt.
__device__ __forceinline__ void async_load_b128(unsigned lds_byte_addr,
                                                const void* gptr) {
  asm volatile("global_load_async_to_lds_b128 %0, %1, off"
               :: "v"(lds_byte_addr), "v"((unsigned long long)(uintptr_t)gptr)
               : "memory");
}
__device__ __forceinline__ void wait_asynccnt0() {
  asm volatile("s_wait_asynccnt 0" ::: "memory");
}

// ---------------------------------------------------------------------------
// Kernel 1: Y = X @ W^T (fp32 in, bf16 out in [H][S][hd] layout).
// grid = (S/64, D/64, 3), block = 128 (4 waves). 64x64 tile per block,
// each wave owns a 32x32 quadrant = 2x2 WMMA tiles. K loop step 32.
// ---------------------------------------------------------------------------
__global__ __launch_bounds__(128)
void qkv_proj(const float* __restrict__ X,
              const float* __restrict__ Wq,
              const float* __restrict__ Wk,
              const float* __restrict__ Wv,
              __bf16* __restrict__ Qb,
              __bf16* __restrict__ Kb,
              __bf16* __restrict__ Vb) {
  __shared__ __align__(16) __bf16 Xs[64][40];   // 64 rows x 32 (+8 pad)
  __shared__ __align__(16) __bf16 Ws[64][40];

  const int tid  = threadIdx.x;
  const int lane = tid & 31;
  const int w    = tid >> 5;
  const int m0   = blockIdx.x * 64;
  const int n0   = blockIdx.y * 64;

  const float* Wm = (blockIdx.z == 0) ? Wq : (blockIdx.z == 1 ? Wk : Wv);
  __bf16*      Y  = (blockIdx.z == 0) ? Qb : (blockIdx.z == 1 ? Kb : Vb);

  const int wm = (w >> 1) * 32;                 // wave row quadrant
  const int wn = (w & 1) * 32;                  // wave col quadrant

  v8f acc00 = {}, acc01 = {}, acc10 = {}, acc11 = {};

  const int lr = tid >> 1;                      // cooperative-load row 0..63
  const int lc = (tid & 1) * 16;                // cols 0 or 16

  for (int kb = 0; kb < D_DIM; kb += 32) {
    __syncthreads();
    // ---- stage X and W tiles (fp32 -> bf16) into LDS ----
    {
      const float4* sx = (const float4*)(X + (m0 + lr) * D_DIM + kb + lc);
      const float4* sw = (const float4*)(Wm + (n0 + lr) * D_DIM + kb + lc);
      v8bf x0, x1, w0, w1;
      float4 f;
      f = sx[0]; x0[0]=(__bf16)f.x; x0[1]=(__bf16)f.y; x0[2]=(__bf16)f.z; x0[3]=(__bf16)f.w;
      f = sx[1]; x0[4]=(__bf16)f.x; x0[5]=(__bf16)f.y; x0[6]=(__bf16)f.z; x0[7]=(__bf16)f.w;
      f = sx[2]; x1[0]=(__bf16)f.x; x1[1]=(__bf16)f.y; x1[2]=(__bf16)f.z; x1[3]=(__bf16)f.w;
      f = sx[3]; x1[4]=(__bf16)f.x; x1[5]=(__bf16)f.y; x1[6]=(__bf16)f.z; x1[7]=(__bf16)f.w;
      f = sw[0]; w0[0]=(__bf16)f.x; w0[1]=(__bf16)f.y; w0[2]=(__bf16)f.z; w0[3]=(__bf16)f.w;
      f = sw[1]; w0[4]=(__bf16)f.x; w0[5]=(__bf16)f.y; w0[6]=(__bf16)f.z; w0[7]=(__bf16)f.w;
      f = sw[2]; w1[0]=(__bf16)f.x; w1[1]=(__bf16)f.y; w1[2]=(__bf16)f.z; w1[3]=(__bf16)f.w;
      f = sw[3]; w1[4]=(__bf16)f.x; w1[5]=(__bf16)f.y; w1[6]=(__bf16)f.z; w1[7]=(__bf16)f.w;
      *(v8bf*)&Xs[lr][lc]     = x0;
      *(v8bf*)&Xs[lr][lc + 8] = x1;
      *(v8bf*)&Ws[lr][lc]     = w0;
      *(v8bf*)&Ws[lr][lc + 8] = w1;
    }
    __syncthreads();

    v16bf a0 = frag_a(&Xs[wm][0],      40);
    v16bf a1 = frag_a(&Xs[wm + 16][0], 40);
    v16bf b0 = frag_bt(&Ws[wn][0],      40);
    v16bf b1 = frag_bt(&Ws[wn + 16][0], 40);
    acc00 = wmma_bf16(a0, b0, acc00);
    acc01 = wmma_bf16(a0, b1, acc01);
    acc10 = wmma_bf16(a1, b0, acc10);
    acc11 = wmma_bf16(a1, b1, acc11);
  }

  // ---- epilogue: C layout m = r + 8*(lane>=16), n = lane&15; bf16 out ----
  const int mrow = m0 + wm + ((lane >> 4) << 3);
  const int ncol = n0 + wn + (lane & 15);
  v8f* accs[4] = { &acc00, &acc01, &acc10, &acc11 };
#pragma unroll
  for (int mt = 0; mt < 2; ++mt) {
#pragma unroll
    for (int nt = 0; nt < 2; ++nt) {
      v8f a = *accs[mt * 2 + nt];
#pragma unroll
      for (int r = 0; r < 8; ++r) {
        int m = mrow + mt * 16 + r;
        int n = ncol + nt * 16;
        Y[((n >> 7) * S_LEN + m) * HD + (n & 127)] = (__bf16)a[r];
      }
    }
  }
}

// ---------------------------------------------------------------------------
// Kernel 2: causal flash attention, bf16 Q/K/V in [H][S][hd] layout.
// grid = (S/128, H), block = 256 (8 waves); each wave owns 16 query rows
// (128-query tile per block).  Per 64-key tile: S = Q*K^T (16 WMMAs/wave),
// online softmax, O += P*V (16 WMMAs/wave).  K tiles staged via async
// global->LDS copies, V via VGPRs with a transposed LDS store; both double
// buffered with a single barrier per tile.  fp32 output [S][D].
// ---------------------------------------------------------------------------
__global__ __launch_bounds__(256)
void attn(const __bf16* __restrict__ Qb,
          const __bf16* __restrict__ Kb,
          const __bf16* __restrict__ Vb,
          float* __restrict__ out) {
  __shared__ __align__(16) __bf16 Ks[2][64][HD];  // [buf][key][hd]   32 KB
  __shared__ __align__(16) __bf16 Vt[2][HD][64];  // [buf][hd][key]   32 KB
  __shared__ __align__(16) __bf16 Pt[128][64];    // probs            16 KB

  const int tid   = threadIdx.x;
  const int lane  = tid & 31;
  const int w     = tid >> 5;                     // wave 0..7
  const int h     = blockIdx.y;
  const int qt    = blockIdx.x;
  const int qbase = qt * 128 + w * 16;            // wave's first query row
  const int ktmax = 2 * qt + 1;                   // last 64-key tile (causal)
  const float scale = 0.08838834764831845f;       // 1/sqrt(128)

  // ---- Q fragments for this wave's 16 rows, kept in registers ----
  v16bf qf[4];
  {
    const __bf16* qrow = Qb + ((h * S_LEN) + qbase + (lane & 15)) * HD;
    const int koff = (lane >> 4) << 3;
#pragma unroll
    for (int kk = 0; kk < 4; ++kk) {
      v8bf lo = *(const v8bf*)(qrow + kk * 32 + koff);
      v8bf hi = *(const v8bf*)(qrow + kk * 32 + 16 + koff);
      v16bf f;
#pragma unroll
      for (int j = 0; j < 8; ++j) { f[j] = lo[j]; f[8 + j] = hi[j]; }
      qf[kk] = f;
    }
  }

  // Cooperative staging geometry (256 threads).
  const int vrow = tid >> 2;                      // V row 0..63
  const int vcb  = (tid & 3) * 32;                // V hd cols {0,32,64,96}
  const __bf16* kgbase = Kb + h * S_LEN * HD;
  const __bf16* vgbase = Vb + h * S_LEN * HD;

  // ---- prologue: stage key tile 0 into buffer 0 ----
  v8bf vreg[4];
  {
    const unsigned lbase = (unsigned)(uintptr_t)&Ks[0][0][0];
#pragma unroll
    for (int i = 0; i < 4; ++i) {
      const unsigned off = (unsigned)(tid + i * 256) * 16u;   // bytes
      async_load_b128(lbase + off, (const char*)kgbase + off);
    }
    const v8bf* vsrc = (const v8bf*)(vgbase + vrow * HD + vcb);
#pragma unroll
    for (int i = 0; i < 4; ++i) vreg[i] = vsrc[i];
#pragma unroll
    for (int i = 0; i < 4; ++i)
#pragma unroll
      for (int e = 0; e < 8; ++e) Vt[0][vcb + i * 8 + e][vrow] = vreg[i][e];
  }

  v8f   o[8];
  float mi[8], li[8];
#pragma unroll
  for (int d = 0; d < 8; ++d) o[d] = (v8f){};
#pragma unroll
  for (int r = 0; r < 8; ++r) { mi[r] = -__builtin_inff(); li[r] = 0.0f; }

  const int ncol = lane & 15;
  const int half = (lane >> 4) << 3;              // 0 or 8 (row offset from r)

  for (int kt = 0; kt <= ktmax; ++kt) {
    const int p  = kt & 1;
    const int kb = kt * 64;
    wait_asynccnt0();                             // own K async slice landed
    __syncthreads();                              // all slices visible; prior
                                                  // reads of buf p^1 done
    const bool havenext = kt < ktmax;
    if (havenext) {
      // stage tile kt+1 into buffer p^1 (K async now, V regs now/LDS later)
      const int nkb = kb + 64;
      const unsigned lbase = (unsigned)(uintptr_t)&Ks[p ^ 1][0][0];
      const char* kg = (const char*)(kgbase + nkb * HD);
#pragma unroll
      for (int i = 0; i < 4; ++i) {
        const unsigned off = (unsigned)(tid + i * 256) * 16u;
        async_load_b128(lbase + off, kg + off);
      }
      const v8bf* vsrc = (const v8bf*)(vgbase + (nkb + vrow) * HD + vcb);
#pragma unroll
      for (int i = 0; i < 4; ++i) vreg[i] = vsrc[i];
    }

    // ---- S = Q * K^T : 4 N-tiles x 4 K-steps ----
    v8f s[4];
#pragma unroll
    for (int nt = 0; nt < 4; ++nt) {
      v8f c = {};
#pragma unroll
      for (int kk = 0; kk < 4; ++kk) {
        v16bf bf = frag_bt(&Ks[p][nt * 16][kk * 32], HD);
        c = wmma_bf16(qf[kk], bf, c);
      }
      s[nt] = c;
    }

    // ---- scale + causal mask ----
#pragma unroll
    for (int nt = 0; nt < 4; ++nt) {
      const int col = kb + nt * 16 + ncol;
#pragma unroll
      for (int r = 0; r < 8; ++r) {
        const int row = qbase + r + half;
        float v = s[nt][r] * scale;
        s[nt][r] = (col <= row) ? v : -__builtin_inff();
      }
    }

    // ---- online softmax (rows in 16-lane halves; reduce across N) ----
#pragma unroll
    for (int r = 0; r < 8; ++r) {
      float rm = fmaxf(fmaxf(s[0][r], s[1][r]), fmaxf(s[2][r], s[3][r]));
      rm = fmaxf(rm, __shfl_xor(rm, 1, 32));
      rm = fmaxf(rm, __shfl_xor(rm, 2, 32));
      rm = fmaxf(rm, __shfl_xor(rm, 4, 32));
      rm = fmaxf(rm, __shfl_xor(rm, 8, 32));
      const float mnew  = fmaxf(mi[r], rm);
      const float alpha = __expf(mi[r] - mnew);
      float rs = 0.0f;
#pragma unroll
      for (int nt = 0; nt < 4; ++nt) {
        float pp = __expf(s[nt][r] - mnew);
        s[nt][r] = pp;
        rs += pp;
      }
      rs += __shfl_xor(rs, 1, 32);
      rs += __shfl_xor(rs, 2, 32);
      rs += __shfl_xor(rs, 4, 32);
      rs += __shfl_xor(rs, 8, 32);
      li[r] = li[r] * alpha + rs;
      mi[r] = mnew;
#pragma unroll
      for (int d = 0; d < 8; ++d) o[d][r] *= alpha;
    }

    // ---- P (C layout) -> LDS bf16 (wave-private strip, LDS is in-order) ----
#pragma unroll
    for (int nt = 0; nt < 4; ++nt)
#pragma unroll
      for (int r = 0; r < 8; ++r)
        Pt[w * 16 + r + half][nt * 16 + ncol] = (__bf16)s[nt][r];

    // ---- O += P * V : 8 hd-tiles x 2 key K-steps ----
    v16bf pf0 = frag_a(&Pt[w * 16][0],  64);
    v16bf pf1 = frag_a(&Pt[w * 16][32], 64);
#pragma unroll
    for (int d = 0; d < 8; ++d) {
      v16bf v0 = frag_bt(&Vt[p][d * 16][0],  64);
      v16bf v1 = frag_bt(&Vt[p][d * 16][32], 64);
      o[d] = wmma_bf16(pf0, v0, o[d]);
      o[d] = wmma_bf16(pf1, v1, o[d]);
    }

    // ---- drop staged V registers into the transposed buffer for kt+1 ----
    if (havenext) {
#pragma unroll
      for (int i = 0; i < 4; ++i)
#pragma unroll
        for (int e = 0; e < 8; ++e)
          Vt[p ^ 1][vcb + i * 8 + e][vrow] = vreg[i][e];
    }
  }

  // ---- finalize: divide by row sum, write fp32 [S][D] ----
#pragma unroll
  for (int r = 0; r < 8; ++r) {
    const float inv = 1.0f / li[r];
    const int row = qbase + r + half;
    float* orow = out + row * D_DIM + h * HD + ncol;
#pragma unroll
    for (int d = 0; d < 8; ++d) orow[d * 16] = o[d][r] * inv;
  }
}

// ---------------------------------------------------------------------------
extern "C" void kernel_launch(void* const* d_in, const int* in_sizes, int n_in,
                              void* d_out, int out_size, void* d_ws, size_t ws_size,
                              hipStream_t stream) {
  (void)in_sizes; (void)n_in; (void)out_size; (void)ws_size;
  const float* X  = (const float*)d_in[0];
  const float* Wq = (const float*)d_in[1];
  const float* Wk = (const float*)d_in[2];
  const float* Wv = (const float*)d_in[3];

  // Workspace: bf16 Q, K, V in [H][S][hd] layout, 8 MB each (24 MB total).
  __bf16* Qb = (__bf16*)d_ws;
  __bf16* Kb = Qb + (size_t)NH * S_LEN * HD;
  __bf16* Vb = Kb + (size_t)NH * S_LEN * HD;

  qkv_proj<<<dim3(S_LEN / 64, D_DIM / 64, 3), 128, 0, stream>>>(
      X, Wq, Wk, Wv, Qb, Kb, Vb);
  attn<<<dim3(S_LEN / 128, NH), 256, 0, stream>>>(
      Qb, Kb, Vb, (float*)d_out);
}
